// SparsityNet_36017595744455
// MI455X (gfx1250) — matrix operand order repaired and checked by reading, not compiled, not run on hardware
//
#include <hip/hip_runtime.h>
#include <math.h>

typedef __attribute__((ext_vector_type(16))) _Float16 v16h;
typedef __attribute__((ext_vector_type(8)))  _Float16 v8h;
typedef __attribute__((ext_vector_type(8)))  float    v8f;

#define DIM   16384
#define NOBS  8192
#define EPSV  1e-20f
#define RCH   32      // row chunks for A^T pass (256 rows each)
#define NB4   64      // partial blocks for DIM-wide sums
#define NB2   64

// ---- ws layout (float offsets) ----
#define OFF_RC    ((size_t)0)
#define OFF_RO    ((size_t)NOBS)
#define OFF_GPART ((size_t)(2*NOBS))
#define OFF_GRAD  (OFF_GPART + (size_t)RCH*DIM)
#define OFF_P8    (OFF_GRAD + (size_t)DIM)
#define OFF_P2    (OFF_P8 + (size_t)NB4*8)
#define OFF_SC    (OFF_P2 + (size_t)NB2*2)
// SC slots: 0-7 alpha, 8 gnorm, 9 dnorm, 10 snorm, 11 gscale, 12 dscale,
// 13 sscale, 14 invL, 15 mL, 16 gmul, 17 m, 18 L, 24..33 stashed sums

__device__ __forceinline__ float softf(float v, float t) {
  float a = fabsf(v) - t;
  a = a > 0.f ? a : 0.f;
  return copysignf(a, v);
}

__device__ __forceinline__ float blockReduce256(float v, float* sm, int tid) {
  sm[tid] = v; __syncthreads();
  for (int s = 128; s > 0; s >>= 1) {
    if (tid < s) sm[tid] += sm[tid + s];
    __syncthreads();
  }
  float r = sm[0]; __syncthreads();
  return r;
}

// ---------------- K1: r_cur/r_old = A@x - b (one streaming pass over A) ----
__global__ __launch_bounds__(256) void k1_resid(
    const float* __restrict__ A, const float* __restrict__ xc,
    const float* __restrict__ xo, const float* __restrict__ bv,
    float* __restrict__ rc, float* __restrict__ ro) {
  __shared__ float sm[256];
  const int tid = threadIdx.x;
  const int row = blockIdx.x;
  const float4* A4  = (const float4*)(A + (size_t)row * DIM);
  const float4* xc4 = (const float4*)xc;
  const float4* xo4 = (const float4*)xo;
  float ac = 0.f, ao = 0.f;
  for (int i = tid; i < DIM / 4; i += 256) {
    float4 a = A4[i], c = xc4[i], o = xo4[i];
    ac += a.x * c.x + a.y * c.y + a.z * c.z + a.w * c.w;
    ao += a.x * o.x + a.y * o.y + a.z * o.z + a.w * o.w;
  }
  float sc_ = blockReduce256(ac, sm, tid);
  float so_ = blockReduce256(ao, sm, tid);
  if (tid == 0) {
    float bb = bv[row];
    rc[row] = sc_ - bb;
    ro[row] = so_ - bb;
  }
}

// ---------------- K2: partial grad = A^T r (second pass over A) -----------
__global__ __launch_bounds__(256) void k2_gradpart(
    const float* __restrict__ A, const float* __restrict__ rc,
    float* __restrict__ gpart) {
  __shared__ float rsh[256];
  const int tid = threadIdx.x;
  const int col = blockIdx.x * 256 + tid;
  const int rbase = blockIdx.y * 256;
  rsh[tid] = rc[rbase + tid];
  __syncthreads();
  const float* Ap = A + (size_t)rbase * DIM + col;
  float acc = 0.f;
#pragma unroll 8
  for (int r = 0; r < 256; ++r)
    acc += Ap[(size_t)r * DIM] * rsh[r];
  gpart[(size_t)blockIdx.y * DIM + col] = acc;
}

// ---------------- K3: grad = sum of partials ------------------------------
__global__ __launch_bounds__(256) void k3_gradsum(
    const float* __restrict__ gpart, float* __restrict__ grad) {
  const int col = blockIdx.x * 256 + threadIdx.x;
  float g = 0.f;
#pragma unroll
  for (int y = 0; y < RCH; ++y) g += gpart[(size_t)y * DIM + col];
  grad[col] = g;
}

// ---------------- K4: 8 elementwise sums over DIM -------------------------
__global__ __launch_bounds__(256) void k4_sums(
    const float* __restrict__ xc, const float* __restrict__ xo,
    const float* __restrict__ grad, float* __restrict__ p8) {
  __shared__ float sm[256];
  const int tid = threadIdx.x;
  const int i = blockIdx.x * 256 + tid;
  float x = xc[i], o = xo[i], g = grad[i];
  float z = (x == 0.f) ? 1.f : 0.f, nz = 1.f - z;
  float d = x - o;
  float acc[8] = { fabsf(x), fabsf(o), d*d*z, d*d*nz, g*g*z, g*g*nz, g*d*z, g*d*nz };
#pragma unroll
  for (int k = 0; k < 8; ++k) {
    float r = blockReduce256(acc[k], sm, tid);
    if (tid == 0) p8[blockIdx.x * 8 + k] = r;
  }
}

// ---------------- K5a: scalar stage 1 (norms, scales) ---------------------
__global__ __launch_bounds__(256) void k5a_scalars(
    const float* __restrict__ rc, const float* __restrict__ ro,
    const float* __restrict__ p8, const float* __restrict__ mu,
    const float* __restrict__ Ls, float* __restrict__ sc) {
  __shared__ float sm[256];
  const int tid = threadIdx.x;
  float a = 0.f, b = 0.f;
  for (int i = tid; i < NOBS; i += 256) {
    float v = rc[i]; a += v * v;
    float w = ro[i]; b += w * w;
  }
  float rc2 = blockReduce256(a, sm, tid);
  float ro2 = blockReduce256(b, sm, tid);
  if (tid == 0) {
    float s[8] = {0,0,0,0,0,0,0,0};
    for (int bk = 0; bk < NB4; ++bk)
      for (int k = 0; k < 8; ++k) s[k] += p8[bk * 8 + k];
    float m = mu[0], L = Ls[0];
    float dnorm = sqrtf(s[2] + s[3]);
    float gnorm = sqrtf(s[4] + s[5]);
    sc[8]  = gnorm;
    sc[9]  = dnorm;
    sc[11] = gnorm > EPSV ? 1.f / gnorm : 1.f;
    sc[12] = dnorm > EPSV ? 1.f / dnorm : 1.f;
    sc[14] = 1.f / L;
    sc[15] = m / L;
    sc[16] = gnorm > EPSV ? 1.f : gnorm;
    sc[17] = m;
    sc[18] = L;
    for (int k = 0; k < 8; ++k) sc[24 + k] = s[k];
    sc[32] = rc2; sc[33] = ro2;
  }
}

// ---------------- K5b: stop-vector partial sums ---------------------------
__global__ __launch_bounds__(256) void k5b_stop(
    const float* __restrict__ xc, const float* __restrict__ grad,
    const float* __restrict__ sc, float* __restrict__ p2) {
  __shared__ float sm[256];
  const int tid = threadIdx.x;
  const int i = blockIdx.x * 256 + tid;
  float gmul = sc[16], invL = sc[14], mL = sc[15];
  float x = xc[i], g = grad[i];
  float z = (x == 0.f) ? 1.f : 0.f, nz = 1.f - z;
  float s = x - softf(x - gmul * g * invL, mL);
  float r0 = blockReduce256(s * s * z, sm, tid);
  float r1 = blockReduce256(s * s * nz, sm, tid);
  if (tid == 0) { p2[blockIdx.x * 2] = r0; p2[blockIdx.x * 2 + 1] = r1; }
}

// ---------------- K5c: features + step-size MLP ---------------------------
__global__ void k5c_feats(
    const float* __restrict__ p2,
    const float* __restrict__ Ws1, const float* __restrict__ Ws2,
    const float* __restrict__ Ws3, const float* __restrict__ Ws4,
    float* __restrict__ sc) {
  if (threadIdx.x != 0) return;
  float sz2 = 0.f, sn2 = 0.f;
  for (int bk = 0; bk < NB2; ++bk) { sz2 += p2[bk * 2]; sn2 += p2[bk * 2 + 1]; }
  float snorm = sqrtf(sz2 + sn2);
  float sscale = snorm > EPSV ? 1.f / snorm : 1.f;
  sc[10] = snorm; sc[13] = sscale;

  float sxc = sc[24], sxo = sc[25], dz2 = sc[26], dn2 = sc[27];
  float gz2 = sc[28], gn2 = sc[29], dotz = sc[30], dotn = sc[31];
  float rc2 = sc[32], ro2 = sc[33];
  float m = sc[17], gnorm = sc[8], dnorm = sc[9], gscale = sc[11], dscale = sc[12];

  float smo = 0.5f * rc2, osmo = 0.5f * ro2;
  float non = m * sxc, onon = m * sxo;
  float loss = smo + non, oloss = osmo + onon;

  float f[12];
  f[0]  = m;
  f[1]  = log1pf(oloss) - log1pf(loss);
  f[2]  = log1pf(osmo) - log1pf(smo);
  f[3]  = log1pf(onon) - log1pf(non);
  f[4]  = log1pf(gnorm * gscale * sqrtf(gz2));
  f[5]  = log1pf(gnorm * gscale * sqrtf(gn2));
  f[6]  = log1pf(dnorm * dscale * sqrtf(dz2));
  f[7]  = log1pf(dnorm * dscale * sqrtf(dn2));
  f[8]  = log1pf(snorm * sscale * sqrtf(sz2));
  f[9]  = log1pf(snorm * sscale * sqrtf(sn2));
  f[10] = dotz * gscale * dscale;
  f[11] = dotn * gscale * dscale;

  float h1[30];
  for (int i = 0; i < 30; ++i) {
    float acc = 0.f;
    for (int j = 0; j < 12; ++j) acc += Ws1[i * 12 + j] * f[j];
    h1[i] = acc > 0.f ? acc : 0.f;
  }
  float h2[20];
  for (int i = 0; i < 20; ++i) {
    float acc = 0.f;
    for (int j = 0; j < 30; ++j) acc += Ws2[i * 30 + j] * h1[j];
    h2[i] = acc > 0.f ? acc : 0.f;
  }
  float h3[10];
  for (int i = 0; i < 10; ++i) {
    float acc = 0.f;
    for (int j = 0; j < 20; ++j) acc += Ws3[i * 20 + j] * h2[j];
    h3[i] = acc > 0.f ? acc : 0.f;
  }
  for (int i = 0; i < 8; ++i) {
    float acc = 0.f;
    for (int j = 0; j < 10; ++j) acc += Ws4[i * 10 + j] * h3[j];
    sc[i] = acc;
  }
}

// ---------------- K6: WMMA channel MLP + final update ---------------------
// Weights staged in LDS as zero-padded 32x32 f16 row-major tiles.
// Activations stored column-major: act[col][k], so B-tile reads and D-tile
// stores are contiguous vector LDS accesses.

// Stage a padded weight matrix (R x Kd) into a 32x32 f16 LDS tile.
__device__ __forceinline__ void stage_w(const float* __restrict__ W,
                                        int R, int Kd, _Float16* dst, int tid) {
  for (int i = tid; i < 32 * 32; i += 256) {
    int m = i >> 5, k = i & 31;
    float w = (m < R && k < Kd) ? W[m * Kd + k] : 0.f;
    dst[i] = (_Float16)w;
  }
}

// 16-bit A-tile (16x32 f16): lane m=lane&15, element e -> K:
//   e=0..15 -> K = kh8 + e (contiguous, kh8 = 16*(lane>>4))
__device__ __forceinline__ v16h load_wtile(const _Float16* wsh, int mtile, int lane) {
  int m   = (lane & 15) + mtile * 16;
  int kh8 = (lane >> 4) * 16;
  v8h lo = *(const v8h*)(wsh + m * 32 + kh8);
  v8h hi = *(const v8h*)(wsh + m * 32 + 8 + kh8);
  v16h a;
#pragma unroll
  for (int e = 0; e < 8; ++e) { a[e] = lo[e]; a[e + 8] = hi[e]; }
  return a;
}

// B-tile (32x16 f16): lanes 0-15 hold K=0..15, lanes 16-31 hold K=16..31,
// column n = lane&15 (+16*ntile). Column-major act -> one 32B contiguous load.
__device__ __forceinline__ v16h load_btile(const _Float16* actw, int ntile, int lane) {
  int n  = (lane & 15) + ntile * 16;
  int kh = (lane >> 4) * 16;
  return *(const v16h*)(actw + n * 32 + kh);
}

// D-tile (16x16 f32): VGPR v, lane -> row M = mh+v, col N = lane&15 (+16*ntile).
// Column-major act -> one 16B contiguous store of 8 converted halves.
__device__ __forceinline__ void store_dtile(_Float16* actw, v8f acc,
                                            int mtile, int ntile, int lane, bool relu) {
  int n  = (lane & 15) + ntile * 16;
  int mh = mtile * 16 + (lane >> 4) * 8;
  v8h outp;
#pragma unroll
  for (int v = 0; v < 8; ++v) {
    float x = acc[v];
    if (relu) x = x > 0.f ? x : 0.f;
    outp[v] = (_Float16)x;
  }
  *(v8h*)(actw + n * 32 + mh) = outp;
}

__device__ __forceinline__ void mlp_layer(const _Float16* wsh,
                                          _Float16* actw, int lane, bool relu, int mtiles) {
  // all B reads before any act writes (in-order DS within a wave)
  v16h b0 = load_btile(actw, 0, lane);
  v16h b1 = load_btile(actw, 1, lane);
  v8f acc[2][2];
  for (int mt = 0; mt < mtiles; ++mt) {
    v16h a = load_wtile(wsh, mt, lane);
    v8f c0 = {};
    v8f c1 = {};
    acc[mt][0] = __builtin_amdgcn_wmma_f32_16x16x32_f16(false, a, false, b0, (short)0, c0, false, false);
    acc[mt][1] = __builtin_amdgcn_wmma_f32_16x16x32_f16(false, a, false, b1, (short)0, c1, false, false);
  }
  for (int mt = 0; mt < mtiles; ++mt) {
    store_dtile(actw, acc[mt][0], mt, 0, lane, relu);
    store_dtile(actw, acc[mt][1], mt, 1, lane, relu);
  }
}

__global__ __launch_bounds__(256) void k6_update(
    const float* __restrict__ xc, const float* __restrict__ xo,
    const float* __restrict__ grad,
    const float* __restrict__ Wu1, const float* __restrict__ Wu2,
    const float* __restrict__ Wu3, const float* __restrict__ Wu4,
    const float* __restrict__ sc, float* __restrict__ out) {
  __shared__ __align__(64) _Float16 act[8][32 * 32];  // per-wave col-major tile
  __shared__ __align__(64) _Float16 w1[32 * 32], w2[32 * 32], w3[32 * 32], w4[32 * 32];
  const int tid  = threadIdx.x;
  const int wave = tid >> 5;
  const int lane = tid & 31;
  const int col  = blockIdx.x * 256 + wave * 32 + lane;

  stage_w(Wu1, 20,  8, w1, tid);
  stage_w(Wu2, 20, 20, w2, tid);
  stage_w(Wu3, 20, 20, w3, tid);
  stage_w(Wu4,  2, 20, w4, tid);

  float al[8];
#pragma unroll
  for (int k = 0; k < 8; ++k) al[k] = sc[k];
  float dnorm = sc[9], gscale = sc[11], dscale = sc[12], sscale = sc[13];
  float invL = sc[14], mL = sc[15], gmul = sc[16];

  float x = xc[col], o = xo[col], g = grad[col];
  float z = (x == 0.f) ? 1.f : 0.f, nz = 1.f - z;
  float d = x - o;
  float gh = g * gscale, dh = d * dscale;
  float s  = x - softf(x - gmul * g * invL, mL);
  float sh = s * sscale;
  float gz = gh * z, gn = gh * nz, sz = sh * z, sn = sh * nz, dz = dh * z, dn = dh * nz;

  float ch[8] = { al[0]*gz, al[1]*gn, al[2]*sz, al[3]*sn,
                  al[4]*dz, al[5]*dn, al[6]*gn*dn, al[7]*gz*dz };
  _Float16* actw = &act[wave][0];
  v16h c0, c1;
#pragma unroll
  for (int e = 0; e < 16; ++e) {
    c0[e] = (e < 8) ? (_Float16)ch[e] : (_Float16)0.f;
    c1[e] = (_Float16)0.f;
  }
  *(v16h*)(actw + lane * 32)      = c0;  // k = 0..15 of this lane's column
  *(v16h*)(actw + lane * 32 + 16) = c1;  // k = 16..31 zero pad

  __syncthreads();  // weights staged

  mlp_layer(w1, actw, lane, true, 2);
  mlp_layer(w2, actw, lane, true, 2);
  mlp_layer(w3, actw, lane, true, 2);
  mlp_layer(w4, actw, lane, false, 1);

  float dir0 = (float)actw[lane * 32 + 0];
  float dir1 = (float)actw[lane * 32 + 1];
  float u = x + (dir0 * z - gmul * g + dnorm * dir1 * nz) * invL;
  out[col] = softf(u, mL);
}

// ------------------------------- launch -----------------------------------
extern "C" void kernel_launch(void* const* d_in, const int* in_sizes, int n_in,
                              void* d_out, int out_size, void* d_ws, size_t ws_size,
                              hipStream_t stream) {
  (void)in_sizes; (void)n_in; (void)out_size; (void)ws_size;
  const float* x_old = (const float*)d_in[0];
  const float* x_cur = (const float*)d_in[1];
  const float* A     = (const float*)d_in[2];
  const float* bv    = (const float*)d_in[3];
  const float* mu    = (const float*)d_in[4];
  const float* Ls    = (const float*)d_in[5];
  const float* Wu1   = (const float*)d_in[6];
  const float* Wu2   = (const float*)d_in[7];
  const float* Wu3   = (const float*)d_in[8];
  const float* Wu4   = (const float*)d_in[9];
  const float* Ws1   = (const float*)d_in[10];
  const float* Ws2   = (const float*)d_in[11];
  const float* Ws3   = (const float*)d_in[12];
  const float* Ws4   = (const float*)d_in[13];
  float* out = (float*)d_out;
  float* ws  = (float*)d_ws;

  float* rc    = ws + OFF_RC;
  float* ro    = ws + OFF_RO;
  float* gpart = ws + OFF_GPART;
  float* grad  = ws + OFF_GRAD;
  float* p8    = ws + OFF_P8;
  float* p2    = ws + OFF_P2;
  float* sc    = ws + OFF_SC;

  k1_resid<<<NOBS, 256, 0, stream>>>(A, x_cur, x_old, bv, rc, ro);
  k2_gradpart<<<dim3(DIM / 256, RCH), 256, 0, stream>>>(A, rc, gpart);
  k3_gradsum<<<DIM / 256, 256, 0, stream>>>(gpart, grad);
  k4_sums<<<NB4, 256, 0, stream>>>(x_cur, x_old, grad, p8);
  k5a_scalars<<<1, 256, 0, stream>>>(rc, ro, p8, mu, Ls, sc);
  k5b_stop<<<NB2, 256, 0, stream>>>(x_cur, grad, sc, p2);
  k5c_feats<<<1, 32, 0, stream>>>(p2, Ws1, Ws2, Ws3, Ws4, sc);
  k6_update<<<DIM / 256, 256, 0, stream>>>(x_cur, x_old, grad,
                                           Wu1, Wu2, Wu3, Wu4, sc, out);
}
